// LanguageCalibration_34024730919225
// MI455X (gfx1250) — matrix operand order repaired and checked by reading, not compiled, not run on hardware
//
#include <hip/hip_runtime.h>

// ---------------------------------------------------------------------------
// MI455X (gfx1250) implementation. All dense GEMMs go through
// v_wmma_f32_16x16x32_bf16 (bf16 operands, f32 accumulate). wave32 everywhere.
// Each wave computes a 16x64 output tile: 4 WMMAs per K-step sharing one A
// fragment. K-step body is split load-phase / compute-phase so the loads for
// the whole step issue as one clause and drain behind the WMMAs.
// Workspace use: ~204 MB (layout in kernel_launch).
// ---------------------------------------------------------------------------

typedef __attribute__((ext_vector_type(16))) __bf16 v16bf;
typedef __attribute__((ext_vector_type(8)))  float  v8f;
typedef __attribute__((ext_vector_type(4)))  float  v4f;

struct GemmCfg {
  long a_bstride, a_sstride; int lda;
  int  wmode;                 // 0: shared W   1: W + b*w_selstride
                              // 2: W + lids[b]*w_selstride
                              // 3: W + (lids[b]*L + lids[jc])*w_selstride
  long w_selstride, w_sstride; int ldw, transB;
  int  bmode;                 // 0: none  1: shared  2: per-language
  long b_selstride;
  long o_bstride, o_sstride; int ldo;
  int  M, N, K;
  float alpha; int relu;
  int  L, jc, zdiv;           // zdiv>1: z = b*zdiv + sub (per-head batching)
};

__device__ __forceinline__ void cvt8(v16bf& dst, int base, v4f a, v4f b)
{
  dst[base + 0] = (__bf16)a[0]; dst[base + 1] = (__bf16)a[1];
  dst[base + 2] = (__bf16)a[2]; dst[base + 3] = (__bf16)a[3];
  dst[base + 4] = (__bf16)b[0]; dst[base + 5] = (__bf16)b[1];
  dst[base + 6] = (__bf16)b[2]; dst[base + 7] = (__bf16)b[3];
}

// One wave per 16x64 output tile. K multiple of 32 (768/512/64 all OK),
// N multiple of 64 (768/512/64/256 all OK). M bounds-checked (M=8 case).
template <int TRANSB>
__global__ __launch_bounds__(32)
void gemm_bf16_wmma(const float* __restrict__ A, const float* __restrict__ W,
                    const float* __restrict__ bias, float* __restrict__ Out,
                    const int* __restrict__ lids, GemmCfg g)
{
  const int lane = threadIdx.x;
  const int half = lane >> 4;       // 0/1
  const int l16  = lane & 15;
  const int n0   = blockIdx.x * 64;
  const int m0   = blockIdx.y * 16;
  const int z    = blockIdx.z;
  const int b    = (g.zdiv > 1) ? (z / g.zdiv) : z;
  const int sub  = (g.zdiv > 1) ? (z % g.zdiv) : 0;

  long aoff = (long)b * g.a_bstride + (long)sub * g.a_sstride;
  long woff;
  if      (g.wmode == 0) woff = 0;
  else if (g.wmode == 1) woff = (long)b * g.w_selstride;
  else if (g.wmode == 2) woff = (long)lids[b] * g.w_selstride;
  else                   woff = (long)(lids[b] * g.L + lids[g.jc]) * g.w_selstride;
  woff += (long)sub * g.w_sstride;
  const long ooff = (long)b * g.o_bstride + (long)sub * g.o_sstride;

  int arow = m0 + l16;                 // A fragment: this lane's M row
  if (arow >= g.M) arow = g.M - 1;     // clamp; garbage rows never stored
  const float* Arow = A + aoff + (long)arow * g.lda + (half << 4);

  // per-subtile W base pointers (all loop-invariant address math hoisted)
  const float* wb[4];
#pragma unroll
  for (int t = 0; t < 4; ++t) {
    if (TRANSB)
      wb[t] = W + woff + (long)(n0 + t * 16 + l16) * g.ldw + (half << 4);
    else
      wb[t] = W + woff + (long)(half << 4) * g.ldw + (n0 + t * 16 + l16);
  }

  v8f acc[4];
#pragma unroll
  for (int t = 0; t < 4; ++t)
    acc[t] = v8f{0.f, 0.f, 0.f, 0.f, 0.f, 0.f, 0.f, 0.f};

  for (int kk = 0; kk < g.K; kk += 32) {
    // ---------------- load phase: issue ALL raw loads for this K-step ------
    v4f a0 = *(const v4f*)(Arow + kk);
    v4f a1 = *(const v4f*)(Arow + kk + 4);
    v4f a2 = *(const v4f*)(Arow + kk + 8);
    v4f a3 = *(const v4f*)(Arow + kk + 12);

    v4f  br[4][4];     // TRANSB path raw data
    float bs[4][16];   // strided path raw data
    if (TRANSB) {
#pragma unroll
      for (int t = 0; t < 4; ++t) {
        const v4f* p = (const v4f*)(wb[t] + kk);
        br[t][0] = p[0]; br[t][1] = p[1]; br[t][2] = p[2]; br[t][3] = p[3];
      }
    } else {
#pragma unroll
      for (int t = 0; t < 4; ++t) {
        const float* p = wb[t] + (long)kk * g.ldw;
#pragma unroll
        for (int e = 0; e < 16; ++e) bs[t][e] = p[(long)e * g.ldw];
      }
    }

    // ---------------- compute phase: converts + 4 WMMAs --------------------
    // A 16x32 bf16 fragment: K(e) = half*16 + e
    v16bf af;
    cvt8(af, 0, a0, a1);
    cvt8(af, 8, a2, a3);
#pragma unroll
    for (int t = 0; t < 4; ++t) {
      // B 32x16 bf16 fragment: K(e) = half*16 + e
      v16bf bfr;
      if (TRANSB) {
        cvt8(bfr, 0, br[t][0], br[t][1]);
        cvt8(bfr, 8, br[t][2], br[t][3]);
      } else {
#pragma unroll
        for (int e = 0; e < 16; ++e) bfr[e] = (__bf16)bs[t][e];
      }
      acc[t] = __builtin_amdgcn_wmma_f32_16x16x32_bf16(
          false, af, false, bfr, (short)0, acc[t], false, false);
    }
  }

#pragma unroll
  for (int t = 0; t < 4; ++t) {
    const int ncol = n0 + t * 16 + l16;
    float bv = 0.f;
    if (g.bmode) {
      long boff = (g.bmode == 2) ? (long)lids[b] * g.b_selstride : 0;
      bv = bias[boff + ncol];
    }
#pragma unroll
    for (int r = 0; r < 8; ++r) {
      int mrow = m0 + (half << 3) + r;   // C/D layout: row = half*8 + r
      if (mrow < g.M) {
        float v = (acc[t][r] + bv) * g.alpha;
        if (g.relu) v = fmaxf(v, 0.f);
        Out[ooff + (long)mrow * g.ldo + ncol] = v;
      }
    }
  }
}

// ---------------------------------------------------------------------------

__device__ __forceinline__ float block_reduce_sum256(float v, float* sm)
{
  int t = threadIdx.x;
  sm[t] = v; __syncthreads();
  for (int off = 128; off > 0; off >>= 1) {
    if (t < off) sm[t] += sm[t + off];
    __syncthreads();
  }
  float r = sm[0]; __syncthreads();
  return r;
}

// per-language LayerNorm + ReLU (adapter). One block per (b,s) row.
__global__ __launch_bounds__(256)
void ln_relu_lang_kernel(const float* __restrict__ X, const float* __restrict__ G,
                         const float* __restrict__ Bm, const int* __restrict__ lids,
                         float* __restrict__ Out, int S, int H)
{
  __shared__ float sm[256];
  long row = blockIdx.x;
  int  lid = lids[(int)(row / S)];
  const float* x = X + row * H;
  float s = 0.f;
  for (int i = threadIdx.x; i < H; i += 256) s += x[i];
  float mean = block_reduce_sum256(s, sm) / (float)H;
  float s2 = 0.f;
  for (int i = threadIdx.x; i < H; i += 256) { float d = x[i] - mean; s2 += d * d; }
  float var = block_reduce_sum256(s2, sm) / (float)H;
  float inv = rsqrtf(var + 1e-5f);
  for (int i = threadIdx.x; i < H; i += 256) {
    float v = (x[i] - mean) * inv * G[(long)lid * H + i] + Bm[(long)lid * H + i];
    Out[row * H + i] = fmaxf(v, 0.f);
  }
}

// residual add + shared LayerNorm -> final aligned_states output
__global__ __launch_bounds__(256)
void ln_residual_kernel(const float* __restrict__ X, const float* __restrict__ R,
                        const float* __restrict__ G, const float* __restrict__ Bm,
                        float* __restrict__ Out, int H)
{
  __shared__ float sm[256];
  long row = blockIdx.x;
  const float* x = X + row * H;
  const float* r = R + row * H;
  float s = 0.f;
  for (int i = threadIdx.x; i < H; i += 256) s += x[i] + r[i];
  float mean = block_reduce_sum256(s, sm) / (float)H;
  float s2 = 0.f;
  for (int i = threadIdx.x; i < H; i += 256) { float d = x[i] + r[i] - mean; s2 += d * d; }
  float var = block_reduce_sum256(s2, sm) / (float)H;
  float inv = rsqrtf(var + 1e-5f);
  for (int i = threadIdx.x; i < H; i += 256)
    Out[row * H + i] = (x[i] + r[i] - mean) * inv * G[i] + Bm[i];
}

// in-place row softmax over the last dim (row length S)
__global__ __launch_bounds__(256)
void softmax_kernel(float* __restrict__ X, int S)
{
  __shared__ float sm[256];
  long row = blockIdx.x;
  float* x = X + row * (long)S;
  int t = threadIdx.x;
  float mx = -1e30f;
  for (int i = t; i < S; i += 256) mx = fmaxf(mx, x[i]);
  sm[t] = mx; __syncthreads();
  for (int off = 128; off > 0; off >>= 1) {
    if (t < off) sm[t] = fmaxf(sm[t], sm[t + off]);
    __syncthreads();
  }
  mx = sm[0]; __syncthreads();
  float s = 0.f;
  for (int i = t; i < S; i += 256) s += __expf(x[i] - mx);
  float inv = 1.f / block_reduce_sum256(s, sm);
  for (int i = t; i < S; i += 256) x[i] = __expf(x[i] - mx) * inv;
}

// alignment_scores = attn.mean(axis=heads)
__global__ void meanheads_kernel(const float* __restrict__ attn, float* __restrict__ out,
                                 long SS, int NH, long total)
{
  long idx = (long)blockIdx.x * blockDim.x + threadIdx.x;
  if (idx >= total) return;
  long b = idx / SS, r = idx % SS;
  float s = 0.f;
  for (int h = 0; h < NH; ++h) s += attn[(b * NH + h) * SS + r];
  out[idx] = s / (float)NH;
}

// P0[b] = align_mats[lids[b], lids[0]]
__global__ void initP_kernel(const float* __restrict__ align, const int* __restrict__ lids,
                             float* __restrict__ P0, long HH, int L, long total)
{
  long idx = (long)blockIdx.x * blockDim.x + threadIdx.x;
  if (idx >= total) return;
  long b = idx / HH, r = idx % HH;
  P0[idx] = align[((long)lids[b] * L + lids[0]) * HH + r];
}

// masked mean pooling over S
__global__ void pooled_kernel(const float* __restrict__ X, const int* __restrict__ mask,
                              float* __restrict__ pooled, int S, int H)
{
  int col = blockIdx.x * blockDim.x + threadIdx.x;
  int b   = blockIdx.y;
  if (col >= H) return;
  float acc = 0.f, ms = 0.f;
  for (int s = 0; s < S; ++s) {
    float m = (float)mask[b * S + s];
    acc += X[((long)b * S + s) * H + col] * m;
    ms  += m;
  }
  pooled[(long)b * H + col] = acc / ms;
}

// row L2-normalize p (C == blockDim == 256)
__global__ __launch_bounds__(256)
void pnorm_kernel(const float* __restrict__ pin, float* __restrict__ pout, int C)
{
  __shared__ float sm[256];
  int b = blockIdx.x, t = threadIdx.x;
  float v  = pin[(long)b * C + t];
  float ss = block_reduce_sum256(v * v, sm);
  pout[(long)b * C + t] = v / sqrtf(ss);
}

// 8x8 contrastive loss, single block of 64 threads
__global__ void loss_kernel(const float* __restrict__ p, const int* __restrict__ lids,
                            const float* __restrict__ temp, float* __restrict__ out_loss,
                            int Bn, int C)
{
  __shared__ float sim[64];
  __shared__ float rowloss[8];
  int t = threadIdx.x;
  int i = t >> 3, j = t & 7;
  float acc = 0.f;
  for (int c = 0; c < C; ++c) acc += p[(long)i * C + c] * p[(long)j * C + c];
  sim[t] = acc / temp[0];
  __syncthreads();
  if (t < Bn) {
    float mx = -1e30f;
    for (int jj = 0; jj < Bn; ++jj) mx = fmaxf(mx, sim[t * 8 + jj]);
    float se = 0.f;
    for (int jj = 0; jj < Bn; ++jj) se += expf(sim[t * 8 + jj] - mx);
    float lse = logf(se);
    float s = 0.f;
    for (int jj = 0; jj < Bn; ++jj) {
      float lab = (lids[t] != lids[jj]) ? 1.f : 0.f;
      s += lab * (sim[t * 8 + jj] - mx - lse);
    }
    rowloss[t] = s;
  }
  __syncthreads();
  if (t == 0) {
    float tot = 0.f;
    for (int ii = 0; ii < Bn; ++ii) tot += rowloss[ii];
    out_loss[0] = -tot / (float)Bn;
  }
}

// ---------------------------------------------------------------------------

static void launch_gemm(hipStream_t stream, const float* A, const float* W,
                        const float* bias, float* Out, const int* lids,
                        const GemmCfg& g, int batches)
{
  dim3 grid((unsigned)(g.N / 64), (unsigned)((g.M + 15) / 16), (unsigned)batches);
  if (g.transB)
    gemm_bf16_wmma<1><<<grid, dim3(32, 1, 1), 0, stream>>>(A, W, bias, Out, lids, g);
  else
    gemm_bf16_wmma<0><<<grid, dim3(32, 1, 1), 0, stream>>>(A, W, bias, Out, lids, g);
}

extern "C" void kernel_launch(void* const* d_in, const int* in_sizes, int n_in,
                              void* d_out, int out_size, void* d_ws, size_t ws_size,
                              hipStream_t stream)
{
  (void)in_sizes; (void)n_in; (void)out_size; (void)ws_size;
  const int B = 8, S = 512, H = 768, L = 5, NH = 12, D = 64, C = 256;
  const long BSH = (long)B * S * H;       // 3,145,728
  const long SS  = (long)S * S;           // 262,144
  const long HH  = (long)H * H;           // 589,824

  const float* x        = (const float*)d_in[0];
  const int*   lids     = (const int*)  d_in[1];
  const int*   mask     = (const int*)  d_in[2];
  const float* adp_w    = (const float*)d_in[3];
  const float* adp_b    = (const float*)d_in[4];
  const float* adp_lng  = (const float*)d_in[5];
  const float* adp_lnb  = (const float*)d_in[6];
  const float* wq_lang  = (const float*)d_in[7];
  const float* bq_lang  = (const float*)d_in[8];
  const float* wk_lang  = (const float*)d_in[9];
  const float* bk_lang  = (const float*)d_in[10];
  const float* attn_wq  = (const float*)d_in[11];
  const float* attn_bq  = (const float*)d_in[12];
  const float* attn_wk  = (const float*)d_in[13];
  const float* attn_bk  = (const float*)d_in[14];
  const float* attn_wv  = (const float*)d_in[15];
  const float* attn_bv  = (const float*)d_in[16];
  const float* attn_wo  = (const float*)d_in[17];
  const float* attn_bo  = (const float*)d_in[18];
  const float* align_m  = (const float*)d_in[19];
  const float* outp_w   = (const float*)d_in[20];
  const float* outp_b   = (const float*)d_in[21];
  const float* ln_g     = (const float*)d_in[22];
  const float* ln_b     = (const float*)d_in[23];
  const float* proj_w1  = (const float*)d_in[24];
  const float* proj_b1  = (const float*)d_in[25];
  const float* proj_w2  = (const float*)d_in[26];
  const float* proj_b2  = (const float*)d_in[27];
  const float* temp     = (const float*)d_in[28];

  // workspace layout (floats): ~53.5M floats ~= 204 MB
  float* ws      = (float*)d_ws;
  float* adapted = ws;                       // BSH
  float* buf1    = adapted + BSH;            // BSH : q_lin -> ctx
  float* buf2    = buf1 + BSH;               // BSH : k_lin -> mha_out
  float* buf3    = buf2 + BSH;               // BSH : q     -> aligned
  float* buf4    = buf3 + BSH;               // BSH : k     -> out (pre-LN)
  float* vbuf    = buf4 + BSH;               // BSH : v
  float* attn    = vbuf + BSH;               // B*NH*S*S
  float* P0      = attn + (long)B * NH * SS; // B*H*H
  float* P1      = P0 + (long)B * HH;        // B*H*H
  float* pooled  = P1 + (long)B * HH;        // B*H
  float* p1buf   = pooled + (long)B * H;     // B*H
  float* pbuf    = p1buf + (long)B * H;      // B*C

  float* out_aligned = (float*)d_out;                 // B*S*H
  float* out_ascore  = out_aligned + BSH;             // B*S*S
  float* out_p       = out_ascore + (long)B * SS;     // B*C
  float* out_loss    = out_p + (long)B * C;           // 1

  GemmCfg g{};

  // --- 1) adapter pre-LN: buf1 = x @ adapter_w[lid] + adapter_b[lid] ---
  g = GemmCfg{};
  g.a_bstride = (long)S * H; g.a_sstride = 0; g.lda = H;
  g.wmode = 2; g.w_selstride = HH; g.w_sstride = 0; g.ldw = H; g.transB = 0;
  g.bmode = 2; g.b_selstride = H;
  g.o_bstride = (long)S * H; g.o_sstride = 0; g.ldo = H;
  g.M = S; g.N = H; g.K = H; g.alpha = 1.f; g.relu = 0;
  g.L = L; g.jc = 0; g.zdiv = 1;
  launch_gemm(stream, x, adp_w, adp_b, buf1, lids, g, B);

  // --- 2) adapted = relu(LN_lang(buf1)) ---
  ln_relu_lang_kernel<<<dim3(B * S), dim3(256), 0, stream>>>(
      buf1, adp_lng, adp_lnb, lids, adapted, S, H);

  // --- 3) q_lin (buf1), k_lin (buf2) ---
  launch_gemm(stream, adapted, wq_lang, bq_lang, buf1, lids, g, B);
  launch_gemm(stream, adapted, wk_lang, bk_lang, buf2, lids, g, B);

  // --- 4) q = (q_lin@Wq + bq)/sqrt(D) (buf3); k (buf4); v (vbuf) ---
  GemmCfg gs = g; gs.wmode = 0; gs.bmode = 1; gs.alpha = 0.125f;
  launch_gemm(stream, buf1, attn_wq, attn_bq, buf3, lids, gs, B);
  gs.alpha = 1.f;
  launch_gemm(stream, buf2, attn_wk, attn_bk, buf4, lids, gs, B);
  launch_gemm(stream, adapted, attn_wv, attn_bv, vbuf, lids, gs, B);

  // --- 5) scores[b,h] = q_h @ k_h^T -> attn ---
  g = GemmCfg{};
  g.a_bstride = (long)S * H; g.a_sstride = D; g.lda = H;
  g.wmode = 1; g.w_selstride = (long)S * H; g.w_sstride = D; g.ldw = H; g.transB = 1;
  g.bmode = 0; g.b_selstride = 0;
  g.o_bstride = (long)NH * SS; g.o_sstride = SS; g.ldo = S;
  g.M = S; g.N = S; g.K = D; g.alpha = 1.f; g.relu = 0;
  g.L = L; g.jc = 0; g.zdiv = NH;
  launch_gemm(stream, buf3, buf4, nullptr, attn, lids, g, B * NH);

  // --- 6) softmax over k; head-average -> alignment_scores output ---
  softmax_kernel<<<dim3((unsigned)(B * NH * S)), dim3(256), 0, stream>>>(attn, S);
  {
    long total = (long)B * SS;
    meanheads_kernel<<<dim3((unsigned)((total + 255) / 256)), dim3(256), 0, stream>>>(
        attn, out_ascore, SS, NH, total);
  }

  // --- 7) ctx[b,:,h*64:] = attn[b,h] @ v_h  (into buf1) ---
  g = GemmCfg{};
  g.a_bstride = (long)NH * SS; g.a_sstride = SS; g.lda = S;
  g.wmode = 1; g.w_selstride = (long)S * H; g.w_sstride = D; g.ldw = H; g.transB = 0;
  g.bmode = 0; g.b_selstride = 0;
  g.o_bstride = (long)S * H; g.o_sstride = D; g.ldo = H;
  g.M = S; g.N = D; g.K = S; g.alpha = 1.f; g.relu = 0;
  g.L = L; g.jc = 0; g.zdiv = NH;
  launch_gemm(stream, attn, vbuf, nullptr, buf1, lids, g, B * NH);

  // --- 8) mha_out = ctx @ Wo + bo (into buf2) ---
  g = GemmCfg{};
  g.a_bstride = (long)S * H; g.a_sstride = 0; g.lda = H;
  g.wmode = 0; g.w_selstride = 0; g.w_sstride = 0; g.ldw = H; g.transB = 0;
  g.bmode = 1; g.b_selstride = 0;
  g.o_bstride = (long)S * H; g.o_sstride = 0; g.ldo = H;
  g.M = S; g.N = H; g.K = H; g.alpha = 1.f; g.relu = 0;
  g.L = L; g.jc = 0; g.zdiv = 1;
  launch_gemm(stream, buf1, attn_wo, attn_bo, buf2, lids, g, B);

  // --- 9) alignment chain: P = M_0 @ M_1 @ ... @ M_7, M_j = A[lid_b, lid_j] ---
  {
    long total = (long)B * HH;
    initP_kernel<<<dim3((unsigned)((total + 255) / 256)), dim3(256), 0, stream>>>(
        align_m, lids, P0, HH, L, total);
  }
  g = GemmCfg{};
  g.a_bstride = HH; g.a_sstride = 0; g.lda = H;
  g.wmode = 3; g.w_selstride = HH; g.w_sstride = 0; g.ldw = H; g.transB = 0;
  g.bmode = 0; g.b_selstride = 0;
  g.o_bstride = HH; g.o_sstride = 0; g.ldo = H;
  g.M = H; g.N = H; g.K = H; g.alpha = 1.f; g.relu = 0;
  g.L = L; g.zdiv = 1;
  {
    float* src = P0; float* dst = P1;
    for (int j = 1; j < B; ++j) {
      g.jc = j;
      launch_gemm(stream, src, align_m, nullptr, dst, lids, g, B);
      float* t = src; src = dst; dst = t;
    }
    // 7 steps starting from P0 => final product lives in P1
  }

  // --- 10) aligned = mha_out @ P_b (into buf3) ---
  g = GemmCfg{};
  g.a_bstride = (long)S * H; g.a_sstride = 0; g.lda = H;
  g.wmode = 1; g.w_selstride = HH; g.w_sstride = 0; g.ldw = H; g.transB = 0;
  g.bmode = 0; g.b_selstride = 0;
  g.o_bstride = (long)S * H; g.o_sstride = 0; g.ldo = H;
  g.M = S; g.N = H; g.K = H; g.alpha = 1.f; g.relu = 0;
  g.L = L; g.jc = 0; g.zdiv = 1;
  launch_gemm(stream, buf2, P1, nullptr, buf3, lids, g, B);

  // --- 11) out = aligned @ out_proj_w + out_proj_b (into buf4) ---
  g.wmode = 0; g.w_selstride = 0; g.bmode = 1;
  launch_gemm(stream, buf3, outp_w, outp_b, buf4, lids, g, B);

  // --- 12) aligned_states = LN(out + adapted) -> d_out ---
  ln_residual_kernel<<<dim3(B * S), dim3(256), 0, stream>>>(
      buf4, adapted, ln_g, ln_b, out_aligned, H);

  // --- 13) masked mean pooling ---
  pooled_kernel<<<dim3((H + 255) / 256, B), dim3(256), 0, stream>>>(
      out_aligned, mask, pooled, S, H);

  // --- 14) p1 = relu(pooled @ W1 + b1) ; p = p1 @ W2 + b2 ---
  g = GemmCfg{};
  g.a_bstride = 0; g.a_sstride = 0; g.lda = H;
  g.wmode = 0; g.w_selstride = 0; g.w_sstride = 0; g.ldw = H; g.transB = 0;
  g.bmode = 1; g.b_selstride = 0;
  g.o_bstride = 0; g.o_sstride = 0; g.ldo = H;
  g.M = B; g.N = H; g.K = H; g.alpha = 1.f; g.relu = 1;
  g.L = L; g.jc = 0; g.zdiv = 1;
  launch_gemm(stream, pooled, proj_w1, proj_b1, p1buf, lids, g, 1);

  g.N = C; g.ldw = C; g.ldo = C; g.relu = 0;
  launch_gemm(stream, p1buf, proj_w2, proj_b2, pbuf, lids, g, 1);

  // --- 15) p normalized -> d_out ; contrastive loss -> d_out ---
  pnorm_kernel<<<dim3(B), dim3(256), 0, stream>>>(pbuf, out_p, C);
  loss_kernel<<<dim3(1), dim3(64), 0, stream>>>(out_p, lids, temp, out_loss, B, C);
}
